// AttentionLayer_66434554135040
// MI455X (gfx1250) — matrix operand order repaired and checked by reading, not compiled, not run on hardware
//
#include <hip/hip_runtime.h>
#include <hip/hip_bf16.h>

// MI455X (gfx1250) fused attention-gate kernel.
//   scores = einsum('bfk,fg->bkg', x, W) + bias ; alpha = softmax(scores, -1)
//   out[b,f,k] = x[b,f,k] * alpha[b,k,f]
// bf16 WMMA (v_wmma_f32_16x16x32_bf16) with fp32 accumulate; W pre-packed to
// bf16 B-fragment order in d_ws. A tiles are DMA'd into double-buffered LDS by
// the Tensor Data Mover (tensor_load_to_lds + s_wait_tensorcnt) so the copy of
// tile i+1 overlaps the 16 WMMAs of tile i.

#define BATCH 64
#define FDIM  1024
#define KDIM  1024
#define MTILE 32      // k-rows per block
#define FSTEP 32      // WMMA K dim for bf16
#define NSTEP (FDIM / FSTEP)

typedef __attribute__((ext_vector_type(16))) __bf16       v16bf;
typedef __attribute__((ext_vector_type(8)))  float        v8f;
typedef __attribute__((ext_vector_type(4)))  unsigned int v4u;
typedef __attribute__((ext_vector_type(8)))  int          v8i;
typedef __attribute__((ext_vector_type(4)))  int          v4i;

union AFrag { v16bf v; __bf16 e[16]; };

#if __has_builtin(__builtin_amdgcn_tensor_load_to_lds) && \
    __has_builtin(__builtin_amdgcn_s_wait_tensorcnt)
#define USE_TDM 1
#else
#define USE_TDM 0
#endif

// ---------------------------------------------------------------------------
// Kernel 0: convert W (fp32 [F][F]) -> bf16 packed [F/32][F][32]:
//   wpk[fb][g][kk] = bf16(W[fb*32+kk][g])  (B fragment = 16 contiguous bf16)
// ---------------------------------------------------------------------------
__global__ __launch_bounds__(256) void pack_weight_bf16(const float* __restrict__ w,
                                                        __bf16* __restrict__ wpk) {
  __shared__ float s[32][65];
  const int fb = blockIdx.x;
  const int g0 = blockIdx.y * 64;
  const int t  = threadIdx.x;
  {
    const int kk = t >> 3;
    const int c0 = (t & 7) * 8;
    const float* src = w + (size_t)(fb * 32 + kk) * FDIM + g0 + c0;
    #pragma unroll
    for (int j = 0; j < 8; ++j) s[kk][c0 + j] = src[j];
  }
  __syncthreads();
  {
    const int g  = t >> 2;
    const int k0 = (t & 3) * 8;
    __bf16* dst = wpk + ((size_t)fb * FDIM + (g0 + g)) * 32 + k0;
    #pragma unroll
    for (int j = 0; j < 8; ++j) dst[j] = (__bf16)s[k0 + j][g];
  }
}

#if USE_TDM
// Issue one TDM 2-D tile DMA: 32 rows (f) x 32 cols (k) of fp32, row stride
// KDIM elements, from gsrc into LDS at lds_off. D# bit layout per ISA ch.8.
__device__ __forceinline__ void tdm_load_tile(const float* gsrc, unsigned lds_off) {
  const unsigned long long ga = (unsigned long long)(uintptr_t)gsrc;
  v4u g0;
  g0[0] = 1u;                                          // count=1 (valid D#)
  g0[1] = lds_off;                                     // lds_addr (bytes)
  g0[2] = (unsigned)(ga & 0xffffffffull);              // global_addr[31:0]
  g0[3] = (unsigned)((ga >> 32) & 0x01ffffffull)       // global_addr[56:32]
        | (2u << 30);                                  // type = 2 ("image")
  v8i g1;
  g1[0] = (int)(2u << 16);                             // data_size = 4 bytes
  g1[1] = (int)((unsigned)(KDIM & 0xffff) << 16);      // tensor_dim0[15:0]
  g1[2] = (int)(((unsigned)KDIM >> 16) |
                ((unsigned)(FDIM & 0xffff) << 16));    // dim0[31:16] | dim1[15:0]
  g1[3] = (int)(((unsigned)FDIM >> 16) |
                ((unsigned)MTILE << 16));              // dim1[31:16] | tile_dim0
  g1[4] = (int)FSTEP;                                  // tile_dim1 (tile_dim2=0)
  g1[5] = (int)KDIM;                                   // tensor_dim0_stride[31:0]
  g1[6] = 0;                                           // stride hi / dim1_stride
  g1[7] = 0;
  const v4i z4 = {0, 0, 0, 0};                         // 2-D tensor: groups 2/3 off
  const v8i z8 = {0, 0, 0, 0, 0, 0, 0, 0};
  __builtin_amdgcn_tensor_load_to_lds(g0, g1, z4, z4, z8, 0);
}
#endif

// ---------------------------------------------------------------------------
// Kernel 1: fused GEMM + softmax + scale.
// ---------------------------------------------------------------------------
__global__ __launch_bounds__(256, 1) void attn_fused(
    const float*  __restrict__ x,      // [B][F][K] fp32
    const __bf16* __restrict__ wpk,    // packed bf16 weight
    const float*  __restrict__ bias,   // [F]
    float*        __restrict__ out) {  // [B][F][K] fp32

  __shared__ float s_x[2][FSTEP][MTILE];           // double-buffered A tile
  __shared__ float s_red[MTILE][8];                // cross-wave row partials
  __shared__ float s_alpha[8][16][MTILE + 1];      // per-wave alpha transpose

  const int tid  = threadIdx.x;
  const int lane = tid & 31;
  const int wave = tid >> 5;
  const int b    = blockIdx.x >> 5;                // 2048 blocks = 64 * 32
  const int k0   = (blockIdx.x & 31) * MTILE;
  const int g0   = wave * 128;                     // 8 waves cover g = 0..1023
  const int gcol = lane & 15;
  const int hi   = lane >> 4;                      // lane half (A/B K split)

  const float* xb = x + (size_t)b * FDIM * KDIM;

  v8f acc[2][8];
  #pragma unroll
  for (int h = 0; h < 2; ++h)
    #pragma unroll
    for (int nt = 0; nt < 8; ++nt)
      #pragma unroll
      for (int r = 0; r < 8; ++r) acc[h][nt][r] = 0.0f;

#if USE_TDM
  if (wave == 0) {
    tdm_load_tile(xb + k0, (unsigned)(uintptr_t)&s_x[0][0][0]);
    __builtin_amdgcn_s_wait_tensorcnt(0);
  }
  __syncthreads();
#else
  const int sx_i = tid >> 3;
  const int sx_j = (tid & 7) * 4;
#endif

  // ---------------- GEMM: sweep the f (reduction) dimension once ----------
  for (int fs = 0; fs < NSTEP; ++fs) {
    const int f0 = fs * FSTEP;
#if USE_TDM
    const int cur = fs & 1;
    if (wave == 0 && fs + 1 < NSTEP)               // overlap DMA with WMMAs
      tdm_load_tile(xb + (size_t)(f0 + FSTEP) * KDIM + k0,
                    (unsigned)(uintptr_t)&s_x[cur ^ 1][0][0]);
    const float (*sx)[MTILE] = s_x[cur];
#else
    __syncthreads();
    {
      const float4 xv = *(const float4*)(xb + (size_t)(f0 + sx_i) * KDIM + k0 + sx_j);
      s_x[0][sx_i][sx_j + 0] = xv.x;
      s_x[0][sx_i][sx_j + 1] = xv.y;
      s_x[0][sx_i][sx_j + 2] = xv.z;
      s_x[0][sx_i][sx_j + 3] = xv.w;
    }
    __syncthreads();
    const float (*sx)[MTILE] = s_x[0];
#endif

    // A fragments (contiguous K order, matching packed B):
    //   lanes 0-15: e[0..15]->K=0..15; lanes16-31: K=16..31. M=lane&15.
    AFrag a[2];
    #pragma unroll
    for (int h = 0; h < 2; ++h) {
      #pragma unroll
      for (int i = 0; i < 16; ++i) {
        const int fi = hi * 16 + i;
        a[h].e[i] = (__bf16)sx[fi][h * 16 + gcol];
      }
    }

    // B fragments: rotate two registers so load(nt+1) overlaps wmma(nt).
    const __bf16* wf = wpk + ((size_t)(f0 >> 5) * FDIM) * 32 + hi * 16;
    v16bf bcur = *(const v16bf*)(wf + (size_t)(g0 + gcol) * 32);
    #pragma unroll
    for (int nt = 0; nt < 8; ++nt) {
      v16bf bnxt = bcur;
      if (nt < 7) bnxt = *(const v16bf*)(wf + (size_t)(g0 + (nt + 1) * 16 + gcol) * 32);
      acc[0][nt] = __builtin_amdgcn_wmma_f32_16x16x32_bf16(
          false, a[0].v, false, bcur, (short)0, acc[0][nt], false, false);
      acc[1][nt] = __builtin_amdgcn_wmma_f32_16x16x32_bf16(
          false, a[1].v, false, bcur, (short)0, acc[1][nt], false, false);
      bcur = bnxt;
    }

#if USE_TDM
    if (wave == 0) __builtin_amdgcn_s_wait_tensorcnt(0);
    __syncthreads();                               // publish next LDS tile
#endif
  }

  // ---------------- bias ---------------------------------------------------
  float bv[8];
  #pragma unroll
  for (int nt = 0; nt < 8; ++nt) bv[nt] = bias[g0 + nt * 16 + gcol];
  #pragma unroll
  for (int h = 0; h < 2; ++h)
    #pragma unroll
    for (int nt = 0; nt < 8; ++nt)
      #pragma unroll
      for (int r = 0; r < 8; ++r) acc[h][nt][r] += bv[nt];

  // ---------------- softmax over g (rows = k) ------------------------------
  // C layout: row k = h*16 + r + 8*hi lives in one 16-lane half at VGPR r.
  #pragma unroll
  for (int h = 0; h < 2; ++h) {
    #pragma unroll
    for (int r = 0; r < 8; ++r) {
      float m = -3.402823466e38f;
      #pragma unroll
      for (int nt = 0; nt < 8; ++nt) m = fmaxf(m, acc[h][nt][r]);
      #pragma unroll
      for (int mk = 1; mk <= 8; mk <<= 1) m = fmaxf(m, __shfl_xor(m, mk, 32));
      if (gcol == 0) s_red[h * 16 + r + hi * 8][wave] = m;
    }
  }
  __syncthreads();
  float rm[2][8];
  #pragma unroll
  for (int h = 0; h < 2; ++h)
    #pragma unroll
    for (int r = 0; r < 8; ++r) {
      float m = -3.402823466e38f;
      #pragma unroll
      for (int wv = 0; wv < 8; ++wv) m = fmaxf(m, s_red[h * 16 + r + hi * 8][wv]);
      rm[h][r] = m;
    }
  __syncthreads();
  #pragma unroll
  for (int h = 0; h < 2; ++h) {
    #pragma unroll
    for (int r = 0; r < 8; ++r) {
      float s = 0.0f;
      #pragma unroll
      for (int nt = 0; nt < 8; ++nt) {
        const float e = __expf(acc[h][nt][r] - rm[h][r]);
        acc[h][nt][r] = e;
        s += e;
      }
      #pragma unroll
      for (int mk = 1; mk <= 8; mk <<= 1) s += __shfl_xor(s, mk, 32);
      if (gcol == 0) s_red[h * 16 + r + hi * 8][wave] = s;
    }
  }
  __syncthreads();
  #pragma unroll
  for (int h = 0; h < 2; ++h)
    #pragma unroll
    for (int r = 0; r < 8; ++r) {
      float s = 0.0f;
      #pragma unroll
      for (int wv = 0; wv < 8; ++wv) s += s_red[h * 16 + r + hi * 8][wv];
      const float inv = 1.0f / s;
      #pragma unroll
      for (int nt = 0; nt < 8; ++nt) acc[h][nt][r] *= inv;
    }

  // ---------------- out[b][g][k0..k0+31] = x * alpha -----------------------
  float* ob = out + (size_t)b * FDIM * KDIM;
  for (int nt = 0; nt < 8; ++nt) {
    __syncthreads();                               // s_alpha reuse guard
    #pragma unroll
    for (int h = 0; h < 2; ++h)
      #pragma unroll
      for (int r = 0; r < 8; ++r)
        s_alpha[wave][gcol][h * 16 + r + 8 * hi] = acc[h][nt][r];  // transpose
    __syncthreads();
    const int gr = lane >> 1;
    const int kc = (lane & 1) * 16;
    const int g  = g0 + nt * 16 + gr;
    const float* src = xb + (size_t)g * KDIM + k0 + kc;
    float*       dst = ob + (size_t)g * KDIM + k0 + kc;
    #pragma unroll
    for (int j = 0; j < 16; j += 4) {
      const float4 xi = *(const float4*)(src + j);
      float4 o;
      o.x = xi.x * s_alpha[wave][gr][kc + j + 0];
      o.y = xi.y * s_alpha[wave][gr][kc + j + 1];
      o.z = xi.z * s_alpha[wave][gr][kc + j + 2];
      o.w = xi.w * s_alpha[wave][gr][kc + j + 3];
      *(float4*)(dst + j) = o;
    }
  }
}

// ---------------------------------------------------------------------------
extern "C" void kernel_launch(void* const* d_in, const int* in_sizes, int n_in,
                              void* d_out, int out_size, void* d_ws, size_t ws_size,
                              hipStream_t stream) {
  const float* x    = (const float*)d_in[0];   // [64,1024,1024] fp32
  const float* w    = (const float*)d_in[1];   // [1024,1024]    fp32
  const float* bias = (const float*)d_in[2];   // [1,1024]       fp32
  float*       out  = (float*)d_out;
  __bf16*      wpk  = (__bf16*)d_ws;           // needs 2 MB scratch

  dim3 gpack(FDIM / 32, FDIM / 64);
  pack_weight_bf16<<<gpack, 256, 0, stream>>>(w, wpk);
  attn_fused<<<BATCH * (KDIM / MTILE), 256, 0, stream>>>(x, wpk, bias, out);
}
